// GrpahWeightRegressor_large_35716948033583
// MI455X (gfx1250) — matrix operand order repaired and checked by reading, not compiled, not run on hardware
//
#include <hip/hip_runtime.h>
#include <hip/hip_bf16.h>
#include <math.h>

// ---------------------------------------------------------------------------
// Problem constants (match reference)
// ---------------------------------------------------------------------------
#define NN      49152
#define DD      512
#define HH      8
#define DH      64
#define EE      131072
#define ETOT    (EE + NN)         // edges + self loops
#define NLAYERS 5
#define BN_EPS  1e-5f
#define NEG_SLOPE 0.2f

// GEMM tiling
#define TILE_M 128
#define TILE_N 128
#define TILE_K 32
#define NITER  (DD / TILE_K)      // 16 K-steps
#define LDS_STRIDE 40             // 40 bf16 = 80B row stride -> conflict-free b128 reads

typedef __bf16 v16bf __attribute__((ext_vector_type(16)));
typedef __bf16 v8bf  __attribute__((ext_vector_type(8)));
typedef float  v8f   __attribute__((ext_vector_type(8)));
typedef float  v4f   __attribute__((ext_vector_type(4)));

// ---------------------------------------------------------------------------
// WMMA fragment loaders (per CDNA5 ISA 7.12.2 layouts, wave32)
//   A 16x32 bf16: lane<16 -> row=lane,   K {0..7, 16..23}
//                 lane>=16 -> row=lane-16, K {8..15, 24..31}
//   B 32x16 bf16: lane<16 -> col=lane,   K {0..7, 16..23}
//                 lane>=16 -> col=lane-16, K {8..15, 24..31}
// ---------------------------------------------------------------------------
__device__ inline v16bf ld_fragA(const __bf16* tile, int row, int lane) {
  int kb = (lane >> 4) * 8;
  const __bf16* b = tile + row * LDS_STRIDE + kb;
  union { v16bf v; v8bf h[2]; } u;
  u.h[0] = *(const v8bf*)(b);
  u.h[1] = *(const v8bf*)(b + 16);
  return u.v;
}
__device__ inline v16bf ld_fragB(const __bf16* tile, int col, int lane) {
  int kb = (lane >> 4) * 8;
  const __bf16* b = tile + col * LDS_STRIDE + kb;
  union { v16bf v; v8bf h[2]; } u;
  u.h[0] = *(const v8bf*)(b);
  u.h[1] = *(const v8bf*)(b + 16);
  return u.v;
}

// ---------------------------------------------------------------------------
// Tile staging: global f32 tiles -> registers -> (hi/lo bf16) -> LDS
// ---------------------------------------------------------------------------
__device__ inline void load_tiles(const float* __restrict__ X,
                                  const float* __restrict__ W,
                                  int m0, int n0, int k0, int tid,
                                  v4f* ar, v4f* br) {
#pragma unroll
  for (int i = 0; i < 4; ++i) {
    int f  = i * 256 + tid;           // 1024 float4s for A (128x32)
    int r  = f >> 3;
    int c4 = f & 7;
    const float* gp = X + (size_t)(m0 + r) * DD + k0 + c4 * 4;
    ar[i] = *(const v4f*)gp;
    // branchless speculative prefetch of the tile after next (wrapped K)
    __builtin_prefetch(X + (size_t)(m0 + r) * DD + ((k0 + 2 * TILE_K) & (DD - 1)) + c4 * 4, 0, 0);
  }
#pragma unroll
  for (int i = 0; i < 4; ++i) {
    int f  = i * 256 + tid;           // 1024 float4s for B (32x128)
    int kk = f >> 5;
    int n4 = f & 31;
    const float* gp = W + (size_t)(k0 + kk) * DD + n0 + n4 * 4;
    br[i] = *(const v4f*)gp;
    __builtin_prefetch(W + (size_t)(((k0 + 2 * TILE_K) & (DD - 1)) + kk) * DD + n0 + n4 * 4, 0, 0);
  }
}

__device__ inline void store_tiles(__bf16* __restrict__ sAh, __bf16* __restrict__ sAl,
                                   __bf16* __restrict__ sBh, __bf16* __restrict__ sBl,
                                   int tid, const v4f* ar, const v4f* br) {
#pragma unroll
  for (int i = 0; i < 4; ++i) {
    int f  = i * 256 + tid;
    int r  = f >> 3;
    int c4 = f & 7;
    __bf16* ph = &sAh[r * LDS_STRIDE + c4 * 4];
    __bf16* pl = &sAl[r * LDS_STRIDE + c4 * 4];
#pragma unroll
    for (int j = 0; j < 4; ++j) {
      float x  = ar[i][j];
      __bf16 h = (__bf16)x;
      ph[j] = h;
      pl[j] = (__bf16)(x - (float)h);
    }
  }
#pragma unroll
  for (int i = 0; i < 4; ++i) {
    int f  = i * 256 + tid;
    int kk = f >> 5;
    int n4 = f & 31;
#pragma unroll
    for (int j = 0; j < 4; ++j) {
      int nn   = n4 * 4 + j;
      float x  = br[i][j];
      __bf16 h = (__bf16)x;
      sBh[nn * LDS_STRIDE + kk] = h;
      sBl[nn * LDS_STRIDE + kk] = (__bf16)(x - (float)h);
    }
  }
}

// ---------------------------------------------------------------------------
// GEMM: Hout[NN x 512] = X[NN x 512] * W[512 x 512]
// split-bf16 (hi/lo) with f32 WMMA accumulation: ~fp32 accuracy, 3 wmma / K=32
// 256 threads (8 waves), tile 128x128, waves 4(M) x 2(N), LDS double-buffered:
// next tile's global loads are in flight while the 24 WMMAs of the current
// tile execute; one s_barrier per K-step.
// ---------------------------------------------------------------------------
__global__ __launch_bounds__(256)
void gat_gemm_bf16x3(const float* __restrict__ X, const float* __restrict__ W,
                     float* __restrict__ Hout) {
  __shared__ __align__(16) __bf16 sAh[2][TILE_M * LDS_STRIDE];
  __shared__ __align__(16) __bf16 sAl[2][TILE_M * LDS_STRIDE];
  __shared__ __align__(16) __bf16 sBh[2][TILE_N * LDS_STRIDE];
  __shared__ __align__(16) __bf16 sBl[2][TILE_N * LDS_STRIDE];

  const int tid  = threadIdx.x;
  const int m0   = blockIdx.y * TILE_M;
  const int n0   = blockIdx.x * TILE_N;
  const int wave = tid >> 5;
  const int lane = tid & 31;
  const int wm   = wave & 3;        // 0..3 -> 32-row slab
  const int wn   = wave >> 2;       // 0..1 -> 64-col slab
  const int lrow = lane & 15;

  v8f acc[2][4] = {};
  v4f ar[4], br[4];

  // prologue: stage tile 0 into buffer 0
  load_tiles(X, W, m0, n0, 0, tid, ar, br);
  store_tiles(sAh[0], sAl[0], sBh[0], sBl[0], tid, ar, br);
  __syncthreads();

  for (int it = 0; it < NITER; ++it) {
    const int cur  = it & 1;
    const int nxt  = cur ^ 1;
    const bool more = (it + 1 < NITER);

    // issue next tile's global loads (overlap with WMMA below)
    if (more) load_tiles(X, W, m0, n0, (it + 1) * TILE_K, tid, ar, br);

    v16bf Ah[2], Al[2], Bh[4], Bl[4];
#pragma unroll
    for (int tm = 0; tm < 2; ++tm) {
      int row = wm * 32 + tm * 16 + lrow;
      Ah[tm] = ld_fragA(sAh[cur], row, lane);
      Al[tm] = ld_fragA(sAl[cur], row, lane);
    }
#pragma unroll
    for (int tn = 0; tn < 4; ++tn) {
      int col = wn * 64 + tn * 16 + lrow;
      Bh[tn] = ld_fragB(sBh[cur], col, lane);
      Bl[tn] = ld_fragB(sBl[cur], col, lane);
    }
#pragma unroll
    for (int tm = 0; tm < 2; ++tm)
#pragma unroll
      for (int tn = 0; tn < 4; ++tn) {
        acc[tm][tn] = __builtin_amdgcn_wmma_f32_16x16x32_bf16(
            false, Ah[tm], false, Bh[tn], (short)0, acc[tm][tn], false, false);
        acc[tm][tn] = __builtin_amdgcn_wmma_f32_16x16x32_bf16(
            false, Ah[tm], false, Bl[tn], (short)0, acc[tm][tn], false, false);
        acc[tm][tn] = __builtin_amdgcn_wmma_f32_16x16x32_bf16(
            false, Al[tm], false, Bh[tn], (short)0, acc[tm][tn], false, false);
      }

    // convert + store next tile into the other buffer (current buffer is
    // being read only by this iteration, which is already done with LDS)
    if (more) store_tiles(sAh[nxt], sAl[nxt], sBh[nxt], sBl[nxt], tid, ar, br);
    __syncthreads();
  }

  // epilogue: C/D layout -> row = base + (lane>=16)*8 + v, col = base + (lane&15)
  const int lhalf = lane >> 4;
#pragma unroll
  for (int tm = 0; tm < 2; ++tm)
#pragma unroll
    for (int tn = 0; tn < 4; ++tn) {
      int row0 = m0 + wm * 32 + tm * 16 + lhalf * 8;
      int col  = n0 + wn * 64 + tn * 16 + lrow;
#pragma unroll
      for (int v = 0; v < 8; ++v)
        Hout[(size_t)(row0 + v) * DD + col] = acc[tm][tn][v];
    }
}

// ---------------------------------------------------------------------------
// Attention logits: als[n,h] = sum_d H[n,h*64+d]*asrc[h,d] (same for ald)
// one wave32 per node; per head all 32 lanes hit the same 64-col window
// ---------------------------------------------------------------------------
__global__ __launch_bounds__(256)
void gat_logits(const float* __restrict__ Hf, const float* __restrict__ asrc,
                const float* __restrict__ adst, float* __restrict__ als,
                float* __restrict__ ald) {
  __shared__ float sa[DD], sd[DD];
  int tid = threadIdx.x;
  sa[tid] = asrc[tid]; sa[tid + 256] = asrc[tid + 256];
  sd[tid] = adst[tid]; sd[tid + 256] = adst[tid + 256];
  __syncthreads();
  int wave = tid >> 5, lane = tid & 31;
  int n = blockIdx.x * 8 + wave;
  const float* hr = Hf + (size_t)n * DD;
#pragma unroll
  for (int h = 0; h < HH; ++h) {
    float ps = 0.f, pd = 0.f;
#pragma unroll
    for (int it = 0; it < 2; ++it) {
      int c = h * DH + it * 32 + lane;
      float v = hr[c];
      ps += v * sa[c];
      pd += v * sd[c];
    }
#pragma unroll
    for (int off = 16; off; off >>= 1) {
      ps += __shfl_xor(ps, off, 32);
      pd += __shfl_xor(pd, off, 32);
    }
    if (lane == 0) { als[n * HH + h] = ps; ald[n * HH + h] = pd; }
  }
}

// ---------------------------------------------------------------------------
// Init kernels (graph-replay safe: everything re-initialized per call)
// ---------------------------------------------------------------------------
__global__ void init_out_bias(float* __restrict__ out, const float* __restrict__ b) {
  int i = blockIdx.x * 256 + threadIdx.x;        // NN*DD threads
  out[i] = b[i & (DD - 1)];
}
__global__ void init_mz(float* __restrict__ m, float* __restrict__ z) {
  int i = blockIdx.x * 256 + threadIdx.x;        // NN*HH threads
  m[i] = -__builtin_huge_valf();
  z[i] = 0.f;
}
__global__ void init_stats(float* __restrict__ s) {
  int i = blockIdx.x * 256 + threadIdx.x;
  if (i < 2 * DD) s[i] = 0.f;
}

// ---------------------------------------------------------------------------
// Edge pass 1: leaky-relu logit + segment max over dst (sign-aware int atomics)
// ---------------------------------------------------------------------------
__device__ inline void edge_sd(const int* __restrict__ ei, int e, int& s, int& d) {
  if (e < EE) { s = ei[e]; d = ei[EE + e]; }
  else        { s = e - EE; d = e - EE; }       // self loops
}

__global__ void edge_logit_max(const int* __restrict__ ei,
                               const float* __restrict__ als,
                               const float* __restrict__ ald,
                               float* __restrict__ logit, float* __restrict__ m) {
  int e = blockIdx.x * 256 + threadIdx.x;
  if (e >= ETOT) return;
  int s, d; edge_sd(ei, e, s, d);
#pragma unroll
  for (int h = 0; h < HH; ++h) {
    float v = als[s * HH + h] + ald[d * HH + h];
    v = (v > 0.f) ? v : NEG_SLOPE * v;
    logit[e * HH + h] = v;
    int* addr = (int*)&m[d * HH + h];
    if (v >= 0.f) atomicMax(addr, __float_as_int(v));
    else          atomicMin((unsigned int*)addr, __float_as_uint(v));
  }
}

// ---------------------------------------------------------------------------
// Edge pass 2: p = exp(logit - max[dst]); segment sum z[dst] (in-place on logit)
// ---------------------------------------------------------------------------
__global__ void edge_exp_sum(const int* __restrict__ ei, const float* __restrict__ m,
                             float* __restrict__ p, float* __restrict__ z) {
  int e = blockIdx.x * 256 + threadIdx.x;
  if (e >= ETOT) return;
  int s, d; edge_sd(ei, e, s, d); (void)s;
#pragma unroll
  for (int h = 0; h < HH; ++h) {
    float pv = expf(p[e * HH + h] - m[d * HH + h]);
    p[e * HH + h] = pv;
    atomicAdd(&z[d * HH + h], pv);
  }
}

// ---------------------------------------------------------------------------
// Edge pass 3: out[dst] += alpha * H[src]; one wave32 per edge (coalesced)
// ---------------------------------------------------------------------------
__global__ __launch_bounds__(256)
void edge_aggregate(const int* __restrict__ ei, const float* __restrict__ p,
                    const float* __restrict__ z, const float* __restrict__ Hf,
                    float* __restrict__ out) {
  int wave = threadIdx.x >> 5, lane = threadIdx.x & 31;
  int e = blockIdx.x * 8 + wave;
  if (e >= ETOT) return;
  int s, d; edge_sd(ei, e, s, d);
  const float* hs = Hf + (size_t)s * DD;
  float* od = out + (size_t)d * DD;
#pragma unroll
  for (int h = 0; h < HH; ++h) {
    float alpha = p[e * HH + h] / (z[d * HH + h] + 1e-16f);
    int c = h * DH + lane;
    atomicAdd(&od[c],      alpha * hs[c]);
    atomicAdd(&od[c + 32], alpha * hs[c + 32]);
  }
}

// ---------------------------------------------------------------------------
// BatchNorm stats: column-wise sum / sumsq, coalesced partials + atomics
// grid = (2, 96), block = 256: thread -> column, blockIdx.y strides rows
// ---------------------------------------------------------------------------
__global__ __launch_bounds__(256)
void bn_stats(const float* __restrict__ out, float* __restrict__ stats) {
  int col = blockIdx.x * 256 + threadIdx.x;
  float s = 0.f, s2 = 0.f;
  for (int r = blockIdx.y; r < NN; r += gridDim.y) {
    float v = out[(size_t)r * DD + col];
    s += v; s2 += v * v;
  }
  atomicAdd(&stats[col], s);
  atomicAdd(&stats[DD + col], s2);
}

__global__ void bn_apply_relu(const float* __restrict__ out,
                              const float* __restrict__ stats,
                              const float* __restrict__ gamma,
                              const float* __restrict__ beta,
                              float* __restrict__ xn) {
  int i = blockIdx.x * 256 + threadIdx.x;      // NN*DD threads
  int c = i & (DD - 1);
  const float inv_n = 1.f / (float)NN;
  float mu  = stats[c] * inv_n;
  float var = stats[DD + c] * inv_n - mu * mu;
  float v = (out[i] - mu) * rsqrtf(var + BN_EPS) * gamma[c] + beta[c];
  xn[i] = (v > 0.f) ? v : 0.f;
}

// ---------------------------------------------------------------------------
// Final head: out[n] = X[n,:] . Wl + bl ; one wave32 per node
// ---------------------------------------------------------------------------
__global__ __launch_bounds__(256)
void final_linear(const float* __restrict__ X, const float* __restrict__ Wl,
                  const float* __restrict__ bl, float* __restrict__ out) {
  __shared__ float sw[DD];
  int tid = threadIdx.x;
  sw[tid] = Wl[tid]; sw[tid + 256] = Wl[tid + 256];
  __syncthreads();
  int wave = tid >> 5, lane = tid & 31;
  int n = blockIdx.x * 8 + wave;
  const float* xr = X + (size_t)n * DD;
  float acc = 0.f;
#pragma unroll
  for (int it = 0; it < 16; ++it) {
    int c = it * 32 + lane;
    acc += xr[c] * sw[c];
  }
#pragma unroll
  for (int off = 16; off; off >>= 1) acc += __shfl_xor(acc, off, 32);
  if (lane == 0) out[n] = acc + bl[0];
}

// ---------------------------------------------------------------------------
// Host-side orchestration
// d_in order: x, edge_index, Ws, Asrc, Adst, Bconv, Gamma, Beta, Wl, bl, bs
// ---------------------------------------------------------------------------
extern "C" void kernel_launch(void* const* d_in, const int* in_sizes, int n_in,
                              void* d_out, int out_size, void* d_ws, size_t ws_size,
                              hipStream_t stream) {
  const float* x     = (const float*)d_in[0];
  const int*   ei    = (const int*)d_in[1];
  const float* Ws    = (const float*)d_in[2];
  const float* Asrc  = (const float*)d_in[3];
  const float* Adst  = (const float*)d_in[4];
  const float* Bconv = (const float*)d_in[5];
  const float* Gamma = (const float*)d_in[6];
  const float* Beta  = (const float*)d_in[7];
  const float* Wl    = (const float*)d_in[8];
  const float* bl    = (const float*)d_in[9];

  const size_t NB  = (size_t)NN * DD * sizeof(float);   // 100.7 MB
  const size_t NH  = (size_t)NN * HH * sizeof(float);   // 1.57 MB
  const size_t EP  = (size_t)ETOT * HH * sizeof(float); // 5.77 MB

  char* ws = (char*)d_ws;
  float* Xb    = (float*)(ws);                 // layer activations (ping)
  float* Hb    = (float*)(ws + NB);            // GEMM output h
  float* Ob    = (float*)(ws + 2 * NB);        // aggregated output
  float* als   = (float*)(ws + 3 * NB);
  float* ald   = (float*)(ws + 3 * NB + NH);
  float* mbuf  = (float*)(ws + 3 * NB + 2 * NH);
  float* zbuf  = (float*)(ws + 3 * NB + 3 * NH);
  float* pbuf  = (float*)(ws + 3 * NB + 4 * NH);
  float* stats = (float*)(ws + 3 * NB + 4 * NH + EP);

  const dim3 gemm_grid(DD / TILE_N, NN / TILE_M);   // (4, 384)
  const int  nd_blocks  = NN * DD / 256;            // 98304
  const int  nh_blocks  = NN * HH / 256;            // 1536
  const int  e_blocks   = (ETOT + 255) / 256;       // 704
  const int  ew_blocks  = (ETOT + 7) / 8;           // wave-per-edge
  const int  nw_blocks  = NN / 8;                   // wave-per-node

  const float* xin = x;
  for (int i = 0; i < NLAYERS; ++i) {
    const float* Wi = Ws + (size_t)i * DD * DD;

    gat_gemm_bf16x3<<<gemm_grid, 256, 0, stream>>>(xin, Wi, Hb);
    gat_logits<<<nw_blocks, 256, 0, stream>>>(Hb, Asrc + (size_t)i * DD,
                                              Adst + (size_t)i * DD, als, ald);
    init_out_bias<<<nd_blocks, 256, 0, stream>>>(Ob, Bconv + (size_t)i * DD);
    init_mz<<<nh_blocks, 256, 0, stream>>>(mbuf, zbuf);
    edge_logit_max<<<e_blocks, 256, 0, stream>>>(ei, als, ald, pbuf, mbuf);
    edge_exp_sum<<<e_blocks, 256, 0, stream>>>(ei, mbuf, pbuf, zbuf);
    edge_aggregate<<<ew_blocks, 256, 0, stream>>>(ei, pbuf, zbuf, Hb, Ob);
    init_stats<<<4, 256, 0, stream>>>(stats);
    bn_stats<<<dim3(2, 96), 256, 0, stream>>>(Ob, stats);
    bn_apply_relu<<<nd_blocks, 256, 0, stream>>>(Ob, stats,
                                                 Gamma + (size_t)i * DD,
                                                 Beta + (size_t)i * DD, Xb);
    xin = Xb;
  }

  final_linear<<<nw_blocks, 256, 0, stream>>>(Xb, Wl, bl, (float*)d_out);
}
